// TGCN_10917806867175
// MI455X (gfx1250) — matrix operand order repaired and checked by reading, not compiled, not run on hardware
//
#include <hip/hip_runtime.h>
#include <hip/hip_bf16.h>
#include <cstdint>
#include <cstddef>

typedef __attribute__((ext_vector_type(16))) _Float16 v16h;
typedef __attribute__((ext_vector_type(8)))  float    v8f;

#define DMODEL 128
#define MBLK   4          // 4 x 16 = 64 rows per block

// ---------------- degree / normalization ----------------

__global__ __launch_bounds__(256) void k_deg_init(float* __restrict__ deg, int n) {
    int i = blockIdx.x * 256 + threadIdx.x;
    if (i < n) deg[i] = 1.0f;                    // self-loop weight
}

__global__ __launch_bounds__(256) void k_deg_scatter(const int* __restrict__ dst,
                                                     const float* __restrict__ ew,
                                                     float* __restrict__ deg, int e) {
    int i = blockIdx.x * 256 + threadIdx.x;
    if (i < e) atomicAdd(&deg[dst[i]], ew[i]);
}

__global__ __launch_bounds__(256) void k_rsqrt(float* __restrict__ d, int n) {
    int i = blockIdx.x * 256 + threadIdx.x;
    if (i < n) { float v = d[i]; d[i] = (v > 0.0f) ? rsqrtf(v) : 0.0f; }
}

// ---------------- WMMA GEMM: out = act(A1@W1 [+ A2@W2] + bias) ----------------
// Block = 256 threads (8 waves) computes a 64-row x 128-col output strip.
// Wave w owns columns [16w,16w+16). B fragments (whole K=128 column strip of W)
// are held in registers (4 x v16h) and reused across 4 M-subtiles whose A
// fragments stream from LDS. 128 v_wmma per block per A pass (16 per wave).

__global__ __launch_bounds__(256) void k_wmma_gemm(
    const float* __restrict__ A1, const float* __restrict__ A2,
    const float* __restrict__ W1, const float* __restrict__ W2,
    const float* __restrict__ bias,
    float* __restrict__ out, int M, int act)
{
    __shared__ _Float16 As[2][MBLK * 16][DMODEL];   // 32 KB
    const int tid  = threadIdx.x;
    const int wave = tid >> 5;
    const int lane = tid & 31;
    const int m0   = blockIdx.x * (MBLK * 16);

    // Stage A tiles (f32 -> f16) into LDS.
    for (int i = tid; i < MBLK * 16 * DMODEL; i += 256) {
        int r = i >> 7, c = i & 127;
        int row = m0 + r;
        As[0][r][c] = (_Float16)((row < M) ? A1[(size_t)row * DMODEL + c] : 0.0f);
        if (A2 != nullptr)
            As[1][r][c] = (_Float16)((row < M) ? A2[(size_t)row * DMODEL + c] : 0.0f);
    }
    __syncthreads();

    const int lh   = lane & 15;      // row (A) / col (B) within tile
    const int ksel = lane >> 4;      // K half selector
    const int ncol = wave * 16 + lh; // output column for this lane

    v8f acc[MBLK];
#pragma unroll
    for (int mt = 0; mt < MBLK; ++mt) acc[mt] = {};

    // ---- A1 @ W1 ----
    {
        v16h bf[4];
#pragma unroll
        for (int ks = 0; ks < 4; ++ks) {
            const int kbb = ks * 32 + ksel * 8;
#pragma unroll
            for (int j = 0; j < 16; ++j)
                bf[ks][j] = (_Float16)W1[(size_t)(kbb + j + (j & 8)) * DMODEL + ncol];
        }
#pragma unroll
        for (int mt = 0; mt < MBLK; ++mt) {
            const int ar = mt * 16 + lh;
#pragma unroll
            for (int ks = 0; ks < 4; ++ks) {
                v16h a;
                const int kb = ks * 32 + ksel * 8;
#pragma unroll
                for (int j = 0; j < 8; ++j) {
                    a[j]     = As[0][ar][kb + j];
                    a[8 + j] = As[0][ar][kb + 16 + j];
                }
                acc[mt] = __builtin_amdgcn_wmma_f32_16x16x32_f16(
                    false, a, false, bf[ks], (short)0, acc[mt], false, false);
            }
        }
    }

    // ---- + A2 @ W2 (concat half) ----
    if (A2 != nullptr) {
        v16h bf[4];
#pragma unroll
        for (int ks = 0; ks < 4; ++ks) {
            const int kbb = ks * 32 + ksel * 8;
#pragma unroll
            for (int j = 0; j < 16; ++j)
                bf[ks][j] = (_Float16)W2[(size_t)(kbb + j + (j & 8)) * DMODEL + ncol];
        }
#pragma unroll
        for (int mt = 0; mt < MBLK; ++mt) {
            const int ar = mt * 16 + lh;
#pragma unroll
            for (int ks = 0; ks < 4; ++ks) {
                v16h a;
                const int kb = ks * 32 + ksel * 8;
#pragma unroll
                for (int j = 0; j < 8; ++j) {
                    a[j]     = As[1][ar][kb + j];
                    a[8 + j] = As[1][ar][kb + 16 + j];
                }
                acc[mt] = __builtin_amdgcn_wmma_f32_16x16x32_f16(
                    false, a, false, bf[ks], (short)0, acc[mt], false, false);
            }
        }
    }

    // Epilogue: bias + activation. D layout: row = mt*16 + ksel*8 + i, col = ncol.
#pragma unroll
    for (int mt = 0; mt < MBLK; ++mt) {
#pragma unroll
        for (int i = 0; i < 8; ++i) {
            int row = m0 + mt * 16 + ksel * 8 + i;
            if (row < M) {
                float v = acc[mt][i];
                if (bias) v += bias[ncol];
                if (act == 1)      v = 1.0f / (1.0f + __expf(-v));   // sigmoid
                else if (act == 2) v = tanhf(v);                     // tanh
                out[(size_t)row * DMODEL + ncol] = v;
            }
        }
    }
}

// ---------------- aggregation ----------------

// C_g = Xw_g * dinv^2 (self loop) + b_g
__global__ __launch_bounds__(256) void k_agg_init(
    const float* __restrict__ Xz, const float* __restrict__ Xr, const float* __restrict__ Xh,
    const float* __restrict__ dinv,
    const float* __restrict__ bz, const float* __restrict__ br, const float* __restrict__ bh,
    float* __restrict__ Cz, float* __restrict__ Cr, float* __restrict__ Ch, int n)
{
    long long t = (long long)blockIdx.x * 256 + threadIdx.x;
    if (t >= (long long)n * DMODEL) return;
    int i = (int)(t >> 7), c = (int)(t & 127);
    float w = dinv[i]; w = w * w;
    size_t idx = (size_t)t;
    Cz[idx] = Xz[idx] * w + bz[c];
    Cr[idx] = Xr[idx] * w + br[c];
    Ch[idx] = Xh[idx] * w + bh[c];
}

// one wave per edge, lane handles 4 channels x 3 gates
__global__ __launch_bounds__(256) void k_edge_scatter(
    const int* __restrict__ src, const int* __restrict__ dst,
    const float* __restrict__ ew, const float* __restrict__ dinv,
    const float* __restrict__ Xz, const float* __restrict__ Xr, const float* __restrict__ Xh,
    float* __restrict__ Cz, float* __restrict__ Cr, float* __restrict__ Ch, int e)
{
    int gw = (blockIdx.x * 256 + threadIdx.x) >> 5;
    int lane = threadIdx.x & 31;
    if (gw >= e) return;
    int s = src[gw], d = dst[gw];
    float norm = dinv[s] * ew[gw] * dinv[d];
    size_t so = (size_t)s * DMODEL + lane * 4;
    size_t dofs = (size_t)d * DMODEL + lane * 4;
    float4 vz = *(const float4*)(Xz + so);
    float4 vr = *(const float4*)(Xr + so);
    float4 vh = *(const float4*)(Xh + so);
    atomicAdd(&Cz[dofs + 0], vz.x * norm);
    atomicAdd(&Cz[dofs + 1], vz.y * norm);
    atomicAdd(&Cz[dofs + 2], vz.z * norm);
    atomicAdd(&Cz[dofs + 3], vz.w * norm);
    atomicAdd(&Cr[dofs + 0], vr.x * norm);
    atomicAdd(&Cr[dofs + 1], vr.y * norm);
    atomicAdd(&Cr[dofs + 2], vr.z * norm);
    atomicAdd(&Cr[dofs + 3], vr.w * norm);
    atomicAdd(&Ch[dofs + 0], vh.x * norm);
    atomicAdd(&Ch[dofs + 1], vh.y * norm);
    atomicAdd(&Ch[dofs + 2], vh.z * norm);
    atomicAdd(&Ch[dofs + 3], vh.w * norm);
}

// ---------------- elementwise tails ----------------

__global__ __launch_bounds__(256) void k_hr(const float* __restrict__ H,
                                            const float* __restrict__ R,
                                            float* __restrict__ HR, long long n) {
    long long t = (long long)blockIdx.x * 256 + threadIdx.x;
    if (t < n) HR[t] = H[t] * R[t];
}

__global__ __launch_bounds__(256) void k_final(const float* __restrict__ Z,
                                               const float* __restrict__ H,
                                               const float* __restrict__ Ht,
                                               float* __restrict__ out, long long n) {
    long long t = (long long)blockIdx.x * 256 + threadIdx.x;
    if (t < n) {
        float z = Z[t];
        out[t] = z * H[t] + (1.0f - z) * Ht[t];
    }
}

// ---------------- launch ----------------

extern "C" void kernel_launch(void* const* d_in, const int* in_sizes, int n_in,
                              void* d_out, int out_size, void* d_ws, size_t ws_size,
                              hipStream_t stream) {
    const float* X    = (const float*)d_in[0];
    const int*   ei   = (const int*)d_in[1];
    const float* ew   = (const float*)d_in[2];
    const float* H    = (const float*)d_in[3];
    const float* W_z  = (const float*)d_in[4];
    const float* b_z  = (const float*)d_in[5];
    const float* Wl_z = (const float*)d_in[6];
    const float* bl_z = (const float*)d_in[7];
    const float* W_r  = (const float*)d_in[8];
    const float* b_r  = (const float*)d_in[9];
    const float* Wl_r = (const float*)d_in[10];
    const float* bl_r = (const float*)d_in[11];
    const float* W_h  = (const float*)d_in[12];
    const float* b_h  = (const float*)d_in[13];
    const float* Wl_h = (const float*)d_in[14];
    const float* bl_h = (const float*)d_in[15];

    const int N = in_sizes[0] / DMODEL;
    const int E = in_sizes[2];
    const int* src = ei;          // edge_index[0]
    const int* dst = ei + E;      // edge_index[1]

    float* ws = (float*)d_ws;
    const size_t nm   = (size_t)N * DMODEL;
    const size_t base = ((size_t)N + 127) & ~(size_t)127;
    float* dinv = ws;
    float* b0 = ws + base;        // Xw_z  -> Z
    float* b1 = b0 + nm;          // Xw_r  -> R
    float* b2 = b1 + nm;          // Xw_h  -> H*R
    float* b3 = b2 + nm;          // Cz    -> H_tilde
    float* b4 = b3 + nm;          // Cr
    float* b5 = b4 + nm;          // Ch

    const int gn  = (N + 255) / 256;
    const int ge  = (E + 255) / 256;
    const int gm  = (N + MBLK * 16 - 1) / (MBLK * 16);
    const long long nmll = (long long)nm;
    const int gnm = (int)((nmll + 255) / 256);
    const int gew = (E + 7) / 8;  // 8 edges (waves) per block

    // normalization
    k_deg_init<<<gn, 256, 0, stream>>>(dinv, N);
    k_deg_scatter<<<ge, 256, 0, stream>>>(dst, ew, dinv, E);
    k_rsqrt<<<gn, 256, 0, stream>>>(dinv, N);

    // Xw = X @ W_g  (WMMA)
    k_wmma_gemm<<<gm, 256, 0, stream>>>(X, nullptr, W_z, nullptr, nullptr, b0, N, 0);
    k_wmma_gemm<<<gm, 256, 0, stream>>>(X, nullptr, W_r, nullptr, nullptr, b1, N, 0);
    k_wmma_gemm<<<gm, 256, 0, stream>>>(X, nullptr, W_h, nullptr, nullptr, b2, N, 0);

    // C_g = self-loop + bias, then edge scatter-add
    k_agg_init<<<gnm, 256, 0, stream>>>(b0, b1, b2, dinv, b_z, b_r, b_h, b3, b4, b5, N);
    k_edge_scatter<<<gew, 256, 0, stream>>>(src, dst, ew, dinv, b0, b1, b2, b3, b4, b5, E);

    // Z = sigmoid([Cz,H] @ Wl_z + bl_z),  R = sigmoid([Cr,H] @ Wl_r + bl_r)
    k_wmma_gemm<<<gm, 256, 0, stream>>>(b3, H, Wl_z, Wl_z + DMODEL * DMODEL, bl_z, b0, N, 1);
    k_wmma_gemm<<<gm, 256, 0, stream>>>(b4, H, Wl_r, Wl_r + DMODEL * DMODEL, bl_r, b1, N, 1);

    // H_tilde = tanh([Ch, H*R] @ Wl_h + bl_h)
    k_hr<<<gnm, 256, 0, stream>>>(H, b1, b2, nmll);
    k_wmma_gemm<<<gm, 256, 0, stream>>>(b5, b2, Wl_h, Wl_h + DMODEL * DMODEL, bl_h, b3, N, 2);

    // out = Z*H + (1-Z)*H_tilde
    k_final<<<gnm, 256, 0, stream>>>(b0, H, b3, (float*)d_out, nmll);
}
